// TransformerBlock_4183298146567
// MI455X (gfx1250) — matrix operand order repaired and checked by reading, not compiled, not run on hardware
//
#include <hip/hip_runtime.h>
#include <hip/hip_bf16.h>

// ---------------------------------------------------------------------------
// Transformer block for MI455X (gfx1250, wave32, WMMA).
// All GEMMs use v_wmma_f32_16x16x32_bf16 (f32 accumulation).
// ---------------------------------------------------------------------------

typedef __bf16 bf16_t;
typedef __bf16 v8bf  __attribute__((ext_vector_type(8)));   // 16 bytes
typedef __bf16 v16bf __attribute__((ext_vector_type(16)));  // 32 bytes
typedef float  v8f   __attribute__((ext_vector_type(8)));

#define B_    2
#define S_    2048
#define D_    1024
#define NH_   16
#define HD_   64
#define M_    4096
#define ROWS_ (B_ * S_)   // 4096

// ---------------------------------------------------------------------------
// Wave-level GEMM: one wave computes a 16(M) x 64(N) tile of D = A * B.
// A: row-major [.., K] bf16.  Bt: B^T row-major, i.e. Bt[n*ldb + k] = B[k][n].
// WMMA bf16 fragment layouts (ISA 7.12.2):
//   A 16x32 : lane L holds row M = L%16, K contiguous from half*16 (16 elems, this code's order)
//   B 32x16 : lane L holds col N = L%16, K contiguous from half*16 (16 elems, this code's order, same as A)
//   C 16x16 : lane L holds col N = L%16, rows M = rr + half*8 (rr = vgpr idx)
// ---------------------------------------------------------------------------
__device__ __forceinline__ void wave_gemm_16x64(
    const bf16_t* __restrict__ A, int lda,
    const bf16_t* __restrict__ Bt, int ldb,
    int K, int m0, int n0, v8f c[4])
{
  const int lane = threadIdx.x & 31;
  const int half = lane >> 4;
  const int r    = lane & 15;
  const bf16_t* arow = A + (size_t)(m0 + r) * lda + half * 16;
  for (int k0 = 0; k0 < K; k0 += 32) {
    union { v16bf v; v8bf h[2]; } au;
    au.h[0] = *(const v8bf*)(arow + k0);
    au.h[1] = *(const v8bf*)(arow + k0 + 8);
#pragma unroll
    for (int j = 0; j < 4; ++j) {
      const bf16_t* bp = Bt + (size_t)(n0 + j * 16 + r) * ldb + k0 + half * 16;
      v16bf b = *(const v16bf*)bp;
      c[j] = __builtin_amdgcn_wmma_f32_16x16x32_bf16(
          false, au.v, false, b, (short)0, c[j], false, false);
    }
  }
}

// ---------------------------------------------------------------------------
// Prep kernels: fp32 weights -> bf16, transposed to K-contiguous (B^T) form.
// ---------------------------------------------------------------------------
__global__ void __launch_bounds__(256) prep_wqkv_kernel(
    const float* __restrict__ w, bf16_t* __restrict__ wt)
{
  size_t i = (size_t)blockIdx.x * 256 + threadIdx.x;   // over 16*192*1024
  if (i >= (size_t)NH_ * 3 * HD_ * D_) return;
  int d = (int)(i % D_);
  int c = (int)((i / D_) % (3 * HD_));
  int n = (int)(i / ((size_t)D_ * 3 * HD_));
  wt[i] = (bf16_t)w[((size_t)n * D_ + d) * (3 * HD_) + c];
}

__global__ void __launch_bounds__(256) prep_w1t_kernel(
    const float* __restrict__ w, bf16_t* __restrict__ wt)
{
  size_t i = (size_t)blockIdx.x * 256 + threadIdx.x;   // wt[m*D + d] = w[d*M + m]
  if (i >= (size_t)M_ * D_) return;
  int d = (int)(i % D_);
  int m = (int)(i / D_);
  wt[i] = (bf16_t)w[(size_t)d * M_ + m];
}

__global__ void __launch_bounds__(256) prep_w2t_kernel(
    const float* __restrict__ w, bf16_t* __restrict__ wt)
{
  size_t i = (size_t)blockIdx.x * 256 + threadIdx.x;   // wt[d*M + m] = w[m*D + d]
  if (i >= (size_t)D_ * M_) return;
  int m = (int)(i % M_);
  int d = (int)(i / M_);
  wt[i] = (bf16_t)w[(size_t)m * D_ + d];
}

__global__ void __launch_bounds__(256) prep_wsum_kernel(
    const float* __restrict__ wout, float* __restrict__ wsum)
{
  int d = blockIdx.x * 256 + threadIdx.x;
  if (d >= D_) return;
  float s = 0.f;
  for (int i = 0; i < NH_ * HD_; ++i) s += wout[(size_t)i * D_ + d];
  wsum[d] = s;
}

// ---------------------------------------------------------------------------
// LayerNorm (unbiased var, eps=1e-4) -> bf16 output
// ---------------------------------------------------------------------------
__global__ void __launch_bounds__(256) ln_kernel(
    const float* __restrict__ x, const float* __restrict__ w,
    const float* __restrict__ b, bf16_t* __restrict__ out)
{
  __shared__ float red[256];
  __shared__ float red2[256];
  int row = blockIdx.x;
  const float* xr = x + (size_t)row * D_;
  float vals[4];
  float s = 0.f, s2 = 0.f;
#pragma unroll
  for (int i = 0; i < 4; ++i) {
    int d = threadIdx.x + i * 256;
    float v = xr[d];
    vals[i] = v; s += v; s2 += v * v;
  }
  red[threadIdx.x] = s; red2[threadIdx.x] = s2;
  __syncthreads();
  for (int st = 128; st > 0; st >>= 1) {
    if ((int)threadIdx.x < st) {
      red[threadIdx.x]  += red[threadIdx.x + st];
      red2[threadIdx.x] += red2[threadIdx.x + st];
    }
    __syncthreads();
  }
  float mu   = red[0] / (float)D_;
  float var  = (red2[0] - (float)D_ * mu * mu) / (float)(D_ - 1);
  float rstd = rsqrtf(var + 1e-4f);
#pragma unroll
  for (int i = 0; i < 4; ++i) {
    int d = threadIdx.x + i * 256;
    out[(size_t)row * D_ + d] = (bf16_t)(w[d] * (vals[i] - mu) * rstd + b[d]);
  }
}

// ---------------------------------------------------------------------------
// QKV GEMM: per (b,n): [S,D] @ [D,192] (Q scaled by 1/8; V written transposed)
// ---------------------------------------------------------------------------
__global__ void __launch_bounds__(128) qkv_kernel(
    const bf16_t* __restrict__ ln1x, const bf16_t* __restrict__ wqkv_t,
    bf16_t* __restrict__ Q, bf16_t* __restrict__ Kb, bf16_t* __restrict__ Vt)
{
  int wave = threadIdx.x >> 5;
  int lane = threadIdx.x & 31;
  int half = lane >> 4, r = lane & 15;
  int bz = blockIdx.z; int b = bz >> 4; int n = bz & 15;
  int m0 = blockIdx.x * 64 + wave * 16;
  int n0 = blockIdx.y * 64;
  const bf16_t* A  = ln1x + (size_t)b * S_ * D_;
  const bf16_t* Bt = wqkv_t + (size_t)n * (3 * HD_) * D_;
  v8f c[4] = {};
  wave_gemm_16x64(A, D_, Bt, D_, D_, m0, n0, c);

  size_t headbase = (size_t)(b * NH_ + n) * S_;
#pragma unroll
  for (int j = 0; j < 4; ++j) {
    int col = n0 + j * 16 + r;
#pragma unroll
    for (int rr = 0; rr < 8; ++rr) {
      int m = m0 + half * 8 + rr;
      float v = c[j][rr];
      if (col < 64) {
        Q[(headbase + m) * HD_ + col] = (bf16_t)(v * 0.125f);     // 1/sqrt(64)
      } else if (col < 128) {
        Kb[(headbase + m) * HD_ + (col - 64)] = (bf16_t)v;
      } else {
        Vt[((size_t)(b * NH_ + n) * HD_ + (col - 128)) * S_ + m] = (bf16_t)v;
      }
    }
  }
}

// ---------------------------------------------------------------------------
// Flash attention (causal, online softmax). One wave = 16 query rows.
// Per-wave LDS staging converts P from C-layout to A-layout (in-order DS,
// no block barrier -> safe with divergent per-wave loop counts).
// ---------------------------------------------------------------------------
__global__ void __launch_bounds__(128) attn_kernel(
    const bf16_t* __restrict__ Q, const bf16_t* __restrict__ Kb,
    const bf16_t* __restrict__ Vt, float* __restrict__ attn_out)
{
  __shared__ __align__(16) bf16_t pbuf[4][16][32];
  int wave = threadIdx.x >> 5;
  int lane = threadIdx.x & 31;
  int half = lane >> 4, r = lane & 15;
  int bn = blockIdx.y;
  int q0 = blockIdx.x * 64 + wave * 16;
  const bf16_t* Qh = Q  + (size_t)bn * S_ * HD_;
  const bf16_t* Kh = Kb + (size_t)bn * S_ * HD_;
  const bf16_t* Vh = Vt + (size_t)bn * HD_ * S_;

  // Q A-fragments for both K=32 steps of HD=64 (reused across all key tiles)
  union { v16bf v; v8bf hh[2]; } aq[2];
  {
    const bf16_t* qrow = Qh + (size_t)(q0 + r) * HD_ + half * 16;
    aq[0].hh[0] = *(const v8bf*)(qrow);
    aq[0].hh[1] = *(const v8bf*)(qrow + 8);
    aq[1].hh[0] = *(const v8bf*)(qrow + 32);
    aq[1].hh[1] = *(const v8bf*)(qrow + 40);
  }

  v8f acc[4] = {};
  float mrow[8], lrow[8];
#pragma unroll
  for (int i = 0; i < 8; ++i) { mrow[i] = -3.0e38f; lrow[i] = 0.f; }

  const int row_q_base = q0 + half * 8;
  const int kt_end = (q0 + 15) / 32;           // inclusive causal tile bound
  for (int kt = 0; kt <= kt_end; ++kt) {
    // scores: 16q x 32k via two 16x16 WMMAs, each over HD in two K=32 steps
    v8f s[2] = {};
#pragma unroll
    for (int kk = 0; kk < 2; ++kk) {
#pragma unroll
      for (int j = 0; j < 2; ++j) {
        const bf16_t* kp = Kh + (size_t)(kt * 32 + j * 16 + r) * HD_ + kk * 32 + half * 16;
        v16bf bk = *(const v16bf*)kp;
        s[j] = __builtin_amdgcn_wmma_f32_16x16x32_bf16(
            false, aq[kk].v, false, bk, (short)0, s[j], false, false);
      }
    }
    // causal mask
#pragma unroll
    for (int j = 0; j < 2; ++j) {
      int colk = kt * 32 + j * 16 + r;
#pragma unroll
      for (int rr = 0; rr < 8; ++rr)
        if (colk > row_q_base + rr) s[j][rr] = -3.0e38f;
    }
    // row max across the 16 lanes of each half-wave
    float tmax[8];
#pragma unroll
    for (int rr = 0; rr < 8; ++rr) tmax[rr] = fmaxf(s[0][rr], s[1][rr]);
#pragma unroll
    for (int off = 1; off < 16; off <<= 1)
#pragma unroll
      for (int rr = 0; rr < 8; ++rr)
        tmax[rr] = fmaxf(tmax[rr], __shfl_xor(tmax[rr], off, 16));

    float scal[8], rsum[8];
#pragma unroll
    for (int rr = 0; rr < 8; ++rr) {
      float mnew = fmaxf(mrow[rr], tmax[rr]);
      scal[rr] = __expf(mrow[rr] - mnew);
      mrow[rr] = mnew;
      float p0 = __expf(s[0][rr] - mnew);
      float p1 = __expf(s[1][rr] - mnew);
      s[0][rr] = p0; s[1][rr] = p1;
      rsum[rr] = p0 + p1;
    }
#pragma unroll
    for (int off = 1; off < 16; off <<= 1)
#pragma unroll
      for (int rr = 0; rr < 8; ++rr)
        rsum[rr] += __shfl_xor(rsum[rr], off, 16);
#pragma unroll
    for (int rr = 0; rr < 8; ++rr) lrow[rr] = lrow[rr] * scal[rr] + rsum[rr];
#pragma unroll
    for (int j2 = 0; j2 < 4; ++j2)
#pragma unroll
      for (int rr = 0; rr < 8; ++rr) acc[j2][rr] *= scal[rr];

    // P: C-layout -> LDS (row-major 16x32) -> A-layout fragment
#pragma unroll
    for (int j = 0; j < 2; ++j)
#pragma unroll
      for (int rr = 0; rr < 8; ++rr)
        pbuf[wave][half * 8 + rr][j * 16 + r] = (bf16_t)s[j][rr];

    union { v16bf v; v8bf hh[2]; } pa;
    pa.hh[0] = *(const v8bf*)&pbuf[wave][r][half * 16];
    pa.hh[1] = *(const v8bf*)&pbuf[wave][r][half * 16 + 8];

    // P(16x32) @ V(32x64): Vt is [HD][S] so B fragments are K-contiguous
#pragma unroll
    for (int j2 = 0; j2 < 4; ++j2) {
      const bf16_t* vp = Vh + (size_t)(j2 * 16 + r) * S_ + kt * 32 + half * 16;
      v16bf bv = *(const v16bf*)vp;
      acc[j2] = __builtin_amdgcn_wmma_f32_16x16x32_bf16(
          false, pa.v, false, bv, (short)0, acc[j2], false, false);
    }
  }

#pragma unroll
  for (int j2 = 0; j2 < 4; ++j2) {
    int hd = j2 * 16 + r;
#pragma unroll
    for (int rr = 0; rr < 8; ++rr) {
      int q = row_q_base + rr;
      attn_out[((size_t)bn * S_ + q) * HD_ + hd] = acc[j2][rr] / lrow[rr];
    }
  }
}

// ---------------------------------------------------------------------------
// Gather (reshape-reinterpret "projection") + residual + LN2
// proj[b,s,d] = attn_out[b, d/64, (d%64)*32 + s/64, s%64] * wsum[d]
// ---------------------------------------------------------------------------
__global__ void __launch_bounds__(256) gather_ln2_kernel(
    const float* __restrict__ x, const float* __restrict__ attn_out,
    const float* __restrict__ wsum, const float* __restrict__ w,
    const float* __restrict__ bias, float* __restrict__ xa,
    bf16_t* __restrict__ ln2x)
{
  __shared__ float red[256];
  __shared__ float red2[256];
  int row = blockIdx.x;               // b*S + s
  int b = row / S_, s = row % S_;
  float vals[4];
  float sm = 0.f, s2 = 0.f;
#pragma unroll
  for (int i = 0; i < 4; ++i) {
    int d = threadIdx.x + i * 256;
    int n      = d >> 6;
    int s_orig = (d & 63) * 32 + (s >> 6);
    int hcol   = s & 63;
    float ao = attn_out[((size_t)(b * NH_ + n) * S_ + s_orig) * HD_ + hcol];
    float v = x[(size_t)row * D_ + d] + ao * wsum[d];
    vals[i] = v; sm += v; s2 += v * v;
    xa[(size_t)row * D_ + d] = v;
  }
  red[threadIdx.x] = sm; red2[threadIdx.x] = s2;
  __syncthreads();
  for (int st = 128; st > 0; st >>= 1) {
    if ((int)threadIdx.x < st) {
      red[threadIdx.x]  += red[threadIdx.x + st];
      red2[threadIdx.x] += red2[threadIdx.x + st];
    }
    __syncthreads();
  }
  float mu   = red[0] / (float)D_;
  float var  = (red2[0] - (float)D_ * mu * mu) / (float)(D_ - 1);
  float rstd = rsqrtf(var + 1e-4f);
#pragma unroll
  for (int i = 0; i < 4; ++i) {
    int d = threadIdx.x + i * 256;
    ln2x[(size_t)row * D_ + d] = (bf16_t)(w[d] * (vals[i] - mu) * rstd + bias[d]);
  }
}

// ---------------------------------------------------------------------------
// MLP: h = gelu(ln2x @ W1 + b1);  out = xa + h @ W2 + b2
// ---------------------------------------------------------------------------
__global__ void __launch_bounds__(128) mlp1_kernel(
    const bf16_t* __restrict__ ln2x, const bf16_t* __restrict__ w1t,
    const float* __restrict__ b1, bf16_t* __restrict__ h)
{
  int wave = threadIdx.x >> 5;
  int lane = threadIdx.x & 31;
  int half = lane >> 4, r = lane & 15;
  int m0 = blockIdx.x * 64 + wave * 16;
  int n0 = blockIdx.y * 64;
  v8f c[4] = {};
  wave_gemm_16x64(ln2x, D_, w1t, D_, D_, m0, n0, c);
#pragma unroll
  for (int j = 0; j < 4; ++j) {
    int col = n0 + j * 16 + r;
    float bb = b1[col];
#pragma unroll
    for (int rr = 0; rr < 8; ++rr) {
      int m = m0 + half * 8 + rr;
      float v = c[j][rr] + bb;
      v = 0.5f * v * (1.0f + erff(v * 0.70710678118654752f));  // exact gelu
      h[(size_t)m * M_ + col] = (bf16_t)v;
    }
  }
}

__global__ void __launch_bounds__(128) mlp2_kernel(
    const bf16_t* __restrict__ h, const bf16_t* __restrict__ w2t,
    const float* __restrict__ b2, const float* __restrict__ xa,
    float* __restrict__ out)
{
  int wave = threadIdx.x >> 5;
  int lane = threadIdx.x & 31;
  int half = lane >> 4, r = lane & 15;
  int m0 = blockIdx.x * 64 + wave * 16;
  int n0 = blockIdx.y * 64;
  v8f c[4] = {};
  wave_gemm_16x64(h, M_, w2t, M_, M_, m0, n0, c);
#pragma unroll
  for (int j = 0; j < 4; ++j) {
    int col = n0 + j * 16 + r;
    float bb = b2[col];
#pragma unroll
    for (int rr = 0; rr < 8; ++rr) {
      int m = m0 + half * 8 + rr;
      out[(size_t)m * D_ + col] = xa[(size_t)m * D_ + col] + c[j][rr] + bb;
    }
  }
}

// ---------------------------------------------------------------------------
// Host launch
// ---------------------------------------------------------------------------
extern "C" void kernel_launch(void* const* d_in, const int* in_sizes, int n_in,
                              void* d_out, int out_size, void* d_ws, size_t ws_size,
                              hipStream_t stream) {
  const float* x     = (const float*)d_in[0];
  const float* W_qkv = (const float*)d_in[1];
  const float* W_out = (const float*)d_in[2];
  const float* ln1_w = (const float*)d_in[3];
  const float* ln1_b = (const float*)d_in[4];
  const float* ln2_w = (const float*)d_in[5];
  const float* ln2_b = (const float*)d_in[6];
  const float* W1    = (const float*)d_in[7];
  const float* b1    = (const float*)d_in[8];
  const float* W2    = (const float*)d_in[9];
  const float* b2    = (const float*)d_in[10];
  float* out = (float*)d_out;

  char* ws = (char*)d_ws;
  size_t off = 0;
  auto alloc = [&](size_t bytes) -> char* {
    char* p = ws + off;
    off += (bytes + 255) & ~(size_t)255;
    return p;
  };

  // persistent weight region (~22.6 MB)
  bf16_t* wqkv_t = (bf16_t*)alloc((size_t)NH_ * 3 * HD_ * D_ * 2);   // 6 MB
  bf16_t* w1t    = (bf16_t*)alloc((size_t)M_ * D_ * 2);              // 8 MB
  bf16_t* w2t    = (bf16_t*)alloc((size_t)D_ * M_ * 2);              // 8 MB
  float*  wsum   = (float*) alloc((size_t)D_ * 4);

  // activation arena; h (32 MB) exactly aliases ln1x+Q+K+Vt (dead by MLP1)
  char* arena = ws + off;
  bf16_t* ln1x = (bf16_t*)alloc((size_t)ROWS_ * D_ * 2);             // 8 MB
  bf16_t* Qb   = (bf16_t*)alloc((size_t)B_ * NH_ * S_ * HD_ * 2);    // 8 MB
  bf16_t* Kbuf = (bf16_t*)alloc((size_t)B_ * NH_ * S_ * HD_ * 2);    // 8 MB
  bf16_t* Vt   = (bf16_t*)alloc((size_t)B_ * NH_ * S_ * HD_ * 2);    // 8 MB
  float*  attn = (float*) alloc((size_t)B_ * NH_ * S_ * HD_ * 4);    // 16 MB
  float*  xa   = (float*) alloc((size_t)ROWS_ * D_ * 4);             // 16 MB
  bf16_t* ln2x = (bf16_t*)alloc((size_t)ROWS_ * D_ * 2);             // 8 MB
  bf16_t* h    = (bf16_t*)arena;                                     // 32 MB alias

  prep_wqkv_kernel<<<(NH_ * 3 * HD_ * D_ + 255) / 256, 256, 0, stream>>>(W_qkv, wqkv_t);
  prep_w1t_kernel <<<(M_ * D_ + 255) / 256, 256, 0, stream>>>(W1, w1t);
  prep_w2t_kernel <<<(D_ * M_ + 255) / 256, 256, 0, stream>>>(W2, w2t);
  prep_wsum_kernel<<<(D_ + 255) / 256, 256, 0, stream>>>(W_out, wsum);

  ln_kernel<<<ROWS_, 256, 0, stream>>>(x, ln1_w, ln1_b, ln1x);

  qkv_kernel<<<dim3(S_ / 64, 3, B_ * NH_), 128, 0, stream>>>(ln1x, wqkv_t, Qb, Kbuf, Vt);

  attn_kernel<<<dim3(S_ / 64, B_ * NH_), 128, 0, stream>>>(Qb, Kbuf, Vt, attn);

  gather_ln2_kernel<<<ROWS_, 256, 0, stream>>>(x, attn, wsum, ln2_w, ln2_b, xa, ln2x);

  mlp1_kernel<<<dim3(ROWS_ / 64, M_ / 64), 128, 0, stream>>>(ln2x, w1t, b1, h);

  mlp2_kernel<<<dim3(ROWS_ / 64, D_ / 64), 128, 0, stream>>>(h, w2t, b2, xa, out);
}
